// VectorCollapseEngine_163208757543
// MI455X (gfx1250) — compile-verified, matrix operand order checked by reading, not run on hardware
//
#include <hip/hip_runtime.h>
#include <hip/hip_bf16.h>
#include <math.h>

#define BB 32768
#define DD 512
#define KK 512
#define NSEG 1536
#define NLAYERS 6

typedef __attribute__((ext_vector_type(16))) __bf16 v16bf;
typedef __attribute__((ext_vector_type(8)))  __bf16 v8bf;
typedef __attribute__((ext_vector_type(8)))  float  v8f;
typedef __attribute__((ext_vector_type(2)))  float  v2f;

__device__ __forceinline__ unsigned short f2bf(float f) {
  unsigned int u = __float_as_uint(f);
  u += 0x7FFFu + ((u >> 16) & 1u);   // round-to-nearest-even
  return (unsigned short)(u >> 16);
}

__device__ __forceinline__ v16bf bf_combine(v8bf lo, v8bf hi) {
  return __builtin_shufflevector(lo, hi, 0,1,2,3,4,5,6,7,8,9,10,11,12,13,14,15);
}

// gfx1250 async copy: global -> LDS, 16 bytes per lane, tracked by ASYNCcnt.
__device__ __forceinline__ void async_b128(unsigned lds_off, unsigned gvoff,
                                           const void* sbase) {
  asm volatile("global_load_async_to_lds_b128 %0, %1, %2 offset:0"
               :: "v"(lds_off), "v"(gvoff), "s"(sbase) : "memory");
}
__device__ __forceinline__ void wait_async_le2() {
  asm volatile("s_wait_asynccnt 0x2" ::: "memory");
}
__device__ __forceinline__ void wait_async_0() {
  asm volatile("s_wait_asynccnt 0x0" ::: "memory");
}

// ---- fp32 -> bf16 bulk convert -----------------------------------------
__global__ void f2bf_kernel(const float* __restrict__ src,
                            unsigned short* __restrict__ dst) {
  int i = blockIdx.x * blockDim.x + threadIdx.x;
  dst[i] = f2bf(src[i]);
}

// ---- weight prep: fp32 [K][N] -> bf16 transposed [N][K] -----------------
__global__ void prep_w(const float* __restrict__ W, unsigned short* __restrict__ Wt) {
  int idx = blockIdx.x * blockDim.x + threadIdx.x;
  int n = idx / DD, k = idx % DD;
  Wt[(size_t)n * DD + k] = f2bf(W[(size_t)k * DD + n]);
}

// ---- routing: argmax_k cos(h_row, centers[label,k]) via f32 WMMA --------
// One wave owns 16 rows; sims vs all 1536 centers; branch-free masked argmax.
__global__ void __launch_bounds__(256) route_kernel(
    const float* __restrict__ H, const float* __restrict__ C,
    const int* __restrict__ labels, int* __restrict__ best) {
  const int lane = threadIdx.x & 31;
  const int wave = (blockIdx.x * blockDim.x + threadIdx.x) >> 5;
  const int rowBase = wave * 16;
  const int half = lane >> 4;
  const int l16 = lane & 15;
  const int rowOff = half * 8;

  int labm[8];
#pragma unroll
  for (int m = 0; m < 8; ++m) labm[m] = labels[rowBase + rowOff + m];

  float bv[8]; int bi[8];
#pragma unroll
  for (int m = 0; m < 8; ++m) { bv[m] = -1e30f; bi[m] = 0; }

  const float* arow = H + (size_t)(rowBase + l16) * DD;
  const v8f vz = {0.f,0.f,0.f,0.f,0.f,0.f,0.f,0.f};

  for (int g = 0; g < 24; ++g) {           // 24 groups x 4 tiles x 16 cols = 1536
    v8f acc[4] = {vz, vz, vz, vz};
    const int colBase = g * 64;
    for (int k = 0; k < DD; k += 4) {
      const int kk = k + half * 2;
      v2f a = *(const v2f*)(arow + kk);
#pragma unroll
      for (int j = 0; j < 4; ++j) {
        const float* brow = C + (size_t)(colBase + j * 16 + l16) * DD;
        v2f b = *(const v2f*)(brow + kk);
        acc[j] = __builtin_amdgcn_wmma_f32_16x16x4_f32(
            false, a, false, b, (short)0, acc[j], false, false);
      }
    }
#pragma unroll
    for (int j = 0; j < 4; ++j) {
      const int col = colBase + j * 16 + l16;
      const int cb = col >> 9;
      const int ci = col & 511;
#pragma unroll
      for (int m = 0; m < 8; ++m) {
        // branch-free: mask to -inf if wrong codebook, then select-update
        float v = (cb == labm[m]) ? acc[j][m] : -1e30f;
        bool better = (v > bv[m]) || ((v == bv[m]) && (ci < bi[m]));
        bv[m] = better ? v : bv[m];
        bi[m] = better ? ci : bi[m];
      }
    }
  }
#pragma unroll
  for (int off = 1; off < 16; off <<= 1) {
#pragma unroll
    for (int m = 0; m < 8; ++m) {
      float ov = __shfl_xor(bv[m], off, 32);
      int   oi = __shfl_xor(bi[m], off, 32);
      bool better = (ov > bv[m]) || ((ov == bv[m]) && (oi < bi[m]));
      bv[m] = better ? ov : bv[m];
      bi[m] = better ? oi : bi[m];
    }
  }
  if (l16 == 0) {
#pragma unroll
    for (int m = 0; m < 8; ++m) best[rowBase + rowOff + m] = bi[m];
  }
}

// ---- GEMM1: T = tanh(Hb @ W1 + b1), bf16 WMMA, async-LDS A staging ------
// Per-wave private double-buffered A tiles: no cross-wave sync in the loop.
__global__ void __launch_bounds__(256) gemm1_kernel(
    const unsigned short* __restrict__ Hb, const unsigned short* __restrict__ W1t,
    const float* __restrict__ b1, unsigned short* __restrict__ T) {
  __shared__ __align__(16) unsigned short Asw[8][2][16 * 40];

  const int tid = threadIdx.x;
  const int lane = tid & 31;
  const int wid = tid >> 5;
  const int wm = wid >> 2;                   // 0..1
  const int wn = wid & 3;                    // 0..3
  const int rowBase = blockIdx.x * 32 + wm * 16;
  const int colBase = blockIdx.y * 256 + wn * 64;
  const int l16 = lane & 15;
  const int half = lane >> 4;

  const v8f vz = {0.f,0.f,0.f,0.f,0.f,0.f,0.f,0.f};
  v8f acc[4] = {vz, vz, vz, vz};

  // async staging geometry: lane -> (row r = lane>>2, 16B chunk c = lane&3)
  const int r = lane >> 2;
  const int c = lane & 3;
  const unsigned ldsBase = (unsigned)(uintptr_t)&Asw[wid][0][0];
  const unsigned ldsLane = (unsigned)(r * 80 + c * 16);      // row stride 80B
  unsigned g = ((unsigned)(rowBase + r) * DD) * 2u + (unsigned)(c * 16);

  // prologue: fill buffer 0
  async_b128(ldsBase + ldsLane, g, Hb);
  async_b128(ldsBase + 640 + ldsLane, g + 8u * DD * 2u, Hb);
  g += 64;

  for (int ks = 0; ks < DD; ks += 32) {
    const int sel = (ks >> 5) & 1;
    if (ks + 32 < DD) {                      // issue next tile, wait for current
      const unsigned nb = ldsBase + (unsigned)((sel ^ 1) * 1280) + ldsLane;
      async_b128(nb, g, Hb);
      async_b128(nb + 640, g + 8u * DD * 2u, Hb);
      g += 64;
      wait_async_le2();
    } else {
      wait_async_0();
    }
    const unsigned short* As = &Asw[wid][sel][0];
    const unsigned short* ap = &As[l16 * 40 + half * 8];
    v16bf a = bf_combine(*(const v8bf*)ap, *(const v8bf*)(ap + 16));
#pragma unroll
    for (int j = 0; j < 4; ++j) {
      const unsigned short* bp = W1t + (size_t)(colBase + j * 16 + l16) * DD + ks + half * 8;
      __builtin_prefetch(bp + 32, 0, 0);     // next k-step of this column
      v16bf b = bf_combine(*(const v8bf*)bp, *(const v8bf*)(bp + 16));
      acc[j] = __builtin_amdgcn_wmma_f32_16x16x32_bf16(
          false, a, false, b, (short)0, acc[j], false, false);
    }
  }
#pragma unroll
  for (int j = 0; j < 4; ++j) {
    const int col = colBase + j * 16 + l16;
    const float bias = b1[col];
#pragma unroll
    for (int m = 0; m < 8; ++m) {
      const int row = rowBase + half * 8 + m;
      T[(size_t)row * DD + col] = f2bf(tanhf(acc[j][m] + bias));
    }
  }
}

// ---- GEMM2 + fused layer update: block owns 16 rows x full 512 cols -----
__global__ void __launch_bounds__(256) gemm2_update_kernel(
    const unsigned short* __restrict__ T, const unsigned short* __restrict__ W2t,
    const float* __restrict__ b2, float* __restrict__ H,
    unsigned short* __restrict__ Hb, const float* __restrict__ C,
    const int* __restrict__ labels, const int* __restrict__ best) {
  __shared__ __align__(16) unsigned short Asw[8][2][16 * 40];
  __shared__ float red[8][16][4];
  __shared__ float s_coef[16];
  __shared__ float s_cap[16];

  const int tid = threadIdx.x;
  const int lane = tid & 31;
  const int wid = tid >> 5;                // wave = 64-col group
  const int rowBase = blockIdx.x * 16;
  const int colBase = wid * 64;
  const int l16 = lane & 15;
  const int half = lane >> 4;
  const int rowOff = half * 8;

  const v8f vz = {0.f,0.f,0.f,0.f,0.f,0.f,0.f,0.f};
  v8f acc[4] = {vz, vz, vz, vz};

  const int r = lane >> 2;
  const int c = lane & 3;
  const unsigned ldsBase = (unsigned)(uintptr_t)&Asw[wid][0][0];
  const unsigned ldsLane = (unsigned)(r * 80 + c * 16);
  unsigned g = ((unsigned)(rowBase + r) * DD) * 2u + (unsigned)(c * 16);

  async_b128(ldsBase + ldsLane, g, T);
  async_b128(ldsBase + 640 + ldsLane, g + 8u * DD * 2u, T);
  g += 64;

  for (int ks = 0; ks < DD; ks += 32) {
    const int sel = (ks >> 5) & 1;
    if (ks + 32 < DD) {
      const unsigned nb = ldsBase + (unsigned)((sel ^ 1) * 1280) + ldsLane;
      async_b128(nb, g, T);
      async_b128(nb + 640, g + 8u * DD * 2u, T);
      g += 64;
      wait_async_le2();
    } else {
      wait_async_0();
    }
    const unsigned short* As = &Asw[wid][sel][0];
    const unsigned short* ap = &As[l16 * 40 + half * 8];
    v16bf a = bf_combine(*(const v8bf*)ap, *(const v8bf*)(ap + 16));
#pragma unroll
    for (int j = 0; j < 4; ++j) {
      const unsigned short* bp = W2t + (size_t)(colBase + j * 16 + l16) * DD + ks + half * 8;
      __builtin_prefetch(bp + 32, 0, 0);
      v16bf b = bf_combine(*(const v8bf*)bp, *(const v8bf*)(bp + 16));
      acc[j] = __builtin_amdgcn_wmma_f32_16x16x32_bf16(
          false, a, false, b, (short)0, acc[j], false, false);
    }
  }

#pragma unroll
  for (int j = 0; j < 4; ++j) {
    const float bias = b2[colBase + j * 16 + l16];
#pragma unroll
    for (int m = 0; m < 8; ++m) acc[j][m] += bias;   // acc = delta
  }

  const float* crow[8];
#pragma unroll
  for (int m = 0; m < 8; ++m) {
    const int row = rowBase + rowOff + m;
    crow[m] = C + (size_t)(labels[row] * KK + best[row]) * DD;
  }

  float hv[4][8], cv[4][8];
  float s_hh[8], s_hc[8], s_dd[8];
#pragma unroll
  for (int m = 0; m < 8; ++m) { s_hh[m] = 0.f; s_hc[m] = 0.f; s_dd[m] = 0.f; }
#pragma unroll
  for (int j = 0; j < 4; ++j) {
    const int col = colBase + j * 16 + l16;
#pragma unroll
    for (int m = 0; m < 8; ++m) {
      const int row = rowBase + rowOff + m;
      float h = H[(size_t)row * DD + col];
      float cc = crow[m][col];
      hv[j][m] = h; cv[j][m] = cc;
      s_hh[m] += h * h;
      s_hc[m] += h * cc;
      float d = h - cc;
      s_dd[m] += d * d;
    }
  }
#pragma unroll
  for (int off = 1; off < 16; off <<= 1) {
#pragma unroll
    for (int m = 0; m < 8; ++m) {
      s_hh[m] += __shfl_xor(s_hh[m], off, 32);
      s_hc[m] += __shfl_xor(s_hc[m], off, 32);
      s_dd[m] += __shfl_xor(s_dd[m], off, 32);
    }
  }
  if (l16 == 0) {
#pragma unroll
    for (int m = 0; m < 8; ++m) {
      red[wid][rowOff + m][0] = s_hh[m];
      red[wid][rowOff + m][1] = s_hc[m];
      red[wid][rowOff + m][2] = s_dd[m];
    }
  }
  __syncthreads();
  if (tid < 16) {
    float hh = 0.f, hc = 0.f, dd = 0.f;
    for (int w = 0; w < 8; ++w) { hh += red[w][tid][0]; hc += red[w][tid][1]; dd += red[w][tid][2]; }
    const int row = rowBase + tid;
    const float strengths[3] = {0.1f, 0.1f, 0.05f};
    float hn = fmaxf(sqrtf(hh), 1e-12f);
    float divg = 0.5f * (1.0f - hc / hn);
    float dn = fmaxf(sqrtf(dd), 1e-12f);
    s_coef[tid] = strengths[labels[row]] * divg / dn;
  }
  __syncthreads();

  float s_nn[8];
#pragma unroll
  for (int m = 0; m < 8; ++m) s_nn[m] = 0.f;
#pragma unroll
  for (int j = 0; j < 4; ++j) {
#pragma unroll
    for (int m = 0; m < 8; ++m) {
      float coef = s_coef[rowOff + m];
      float hnew = hv[j][m] + acc[j][m] - coef * (hv[j][m] - cv[j][m]);
      acc[j][m] = hnew;
      s_nn[m] += hnew * hnew;
    }
  }
#pragma unroll
  for (int off = 1; off < 16; off <<= 1) {
#pragma unroll
    for (int m = 0; m < 8; ++m) s_nn[m] += __shfl_xor(s_nn[m], off, 32);
  }
  if (l16 == 0) {
#pragma unroll
    for (int m = 0; m < 8; ++m) red[wid][rowOff + m][0] = s_nn[m];
  }
  __syncthreads();
  if (tid < 16) {
    float nn = 0.f;
    for (int w = 0; w < 8; ++w) nn += red[w][tid][0];
    float n = sqrtf(nn);
    s_cap[tid] = (n > 10.0f) ? (10.0f / (n + 1e-8f)) : 1.0f;
  }
  __syncthreads();
#pragma unroll
  for (int j = 0; j < 4; ++j) {
    const int col = colBase + j * 16 + l16;
#pragma unroll
    for (int m = 0; m < 8; ++m) {
      const int row = rowBase + rowOff + m;
      float hc2 = acc[j][m] * s_cap[rowOff + m];
      H[(size_t)row * DD + col] = hc2;
      Hb[(size_t)row * DD + col] = f2bf(hc2);   // bf16 shadow for next GEMM1
    }
  }
}

// ---- segment sum helpers ------------------------------------------------
__global__ void zero_kernel(float* p, int n) {
  int i = blockIdx.x * blockDim.x + threadIdx.x;
  if (i < n) p[i] = 0.0f;
}

__global__ void __launch_bounds__(256) accum_kernel(
    const float* __restrict__ H, const int* __restrict__ labels,
    const int* __restrict__ best2, float* __restrict__ ssum,
    float* __restrict__ cnt) {
  __shared__ float rbuf[256];
  const int row = blockIdx.x;
  const int tid = threadIdx.x;
  const float* hr = H + (size_t)row * DD;
  float2 hval = *(const float2*)(hr + tid * 2);
  rbuf[tid] = hval.x * hval.x + hval.y * hval.y;
  __syncthreads();
  for (int s = 128; s > 0; s >>= 1) {
    if (tid < s) rbuf[tid] += rbuf[tid + s];
    __syncthreads();
  }
  float hn = fmaxf(sqrtf(rbuf[0]), 1e-12f);
  const int seg = labels[row] * KK + best2[row];
  float* dst = ssum + (size_t)seg * DD + tid * 2;
  atomicAdd(dst, hval.x / hn);
  atomicAdd(dst + 1, hval.y / hn);
  if (tid == 0) atomicAdd(&cnt[seg], 1.0f);
}

__global__ void __launch_bounds__(256) ema_kernel(
    const float* __restrict__ C, const float* __restrict__ ssum,
    const float* __restrict__ cnt, float* __restrict__ outC) {
  __shared__ float rbuf[256];
  const int s = blockIdx.x;
  const int tid = threadIdx.x;
  const float cn = cnt[s];
  const float denom = fmaxf(cn, 1.0f);
  float2 oldv = *(const float2*)(C + (size_t)s * DD + tid * 2);
  float2 sv = *(const float2*)(ssum + (size_t)s * DD + tid * 2);
  float u0 = 0.95f * oldv.x + 0.05f * (sv.x / denom);
  float u1 = 0.95f * oldv.y + 0.05f * (sv.y / denom);
  rbuf[tid] = u0 * u0 + u1 * u1;
  __syncthreads();
  for (int st = 128; st > 0; st >>= 1) {
    if (tid < st) rbuf[tid] += rbuf[tid + st];
    __syncthreads();
  }
  float un = fmaxf(sqrtf(rbuf[0]), 1e-12f);
  float* dst = outC + (size_t)s * DD + tid * 2;
  dst[0] = (cn > 0.f) ? u0 / un : oldv.x;
  dst[1] = (cn > 0.f) ? u1 / un : oldv.y;
}

extern "C" void kernel_launch(void* const* d_in, const int* in_sizes, int n_in,
                              void* d_out, int out_size, void* d_ws, size_t ws_size,
                              hipStream_t stream) {
  (void)in_sizes; (void)n_in; (void)out_size; (void)ws_size;
  const float* h0      = (const float*)d_in[0];
  const int*   labels  = (const int*)d_in[1];
  const float* centers = (const float*)d_in[2];
  const float* W1      = (const float*)d_in[3];
  const float* b1      = (const float*)d_in[4];
  const float* W2      = (const float*)d_in[5];
  const float* b2      = (const float*)d_in[6];
  float* out_h = (float*)d_out;
  float* out_c = out_h + (size_t)BB * DD;

  char* ws = (char*)d_ws;
  float* H            = (float*)ws;          ws += (size_t)BB * DD * 4;
  unsigned short* Hb  = (unsigned short*)ws; ws += (size_t)BB * DD * 2;
  unsigned short* T   = (unsigned short*)ws; ws += (size_t)BB * DD * 2;
  unsigned short* W1t = (unsigned short*)ws; ws += (size_t)DD * DD * 2;
  unsigned short* W2t = (unsigned short*)ws; ws += (size_t)DD * DD * 2;
  int* best           = (int*)ws;            ws += (size_t)BB * 4;
  int* best2          = (int*)ws;            ws += (size_t)BB * 4;
  float* ssum         = (float*)ws;          ws += (size_t)NSEG * DD * 4;
  float* cnt          = (float*)ws;          ws += (size_t)NSEG * 4;

  hipMemcpyAsync(H, h0, (size_t)BB * DD * 4, hipMemcpyDeviceToDevice, stream);
  f2bf_kernel<<<(BB * DD) / 256, 256, 0, stream>>>(h0, Hb);
  prep_w<<<(DD * DD) / 256, 256, 0, stream>>>(W1, W1t);
  prep_w<<<(DD * DD) / 256, 256, 0, stream>>>(W2, W2t);

  route_kernel<<<BB / 16 / 8, 256, 0, stream>>>(H, centers, labels, best);

  for (int l = 0; l < NLAYERS; ++l) {
    gemm1_kernel<<<dim3(BB / 32, 2), 256, 0, stream>>>(Hb, W1t, b1, T);
    gemm2_update_kernel<<<BB / 16, 256, 0, stream>>>(T, W2t, b2, H, Hb,
                                                     centers, labels, best);
  }

  route_kernel<<<BB / 16 / 8, 256, 0, stream>>>(H, centers, labels, best2);

  int zn = NSEG * DD + NSEG;   // ssum and cnt are contiguous in ws
  zero_kernel<<<(zn + 255) / 256, 256, 0, stream>>>(ssum, zn);
  accum_kernel<<<BB, 256, 0, stream>>>(H, labels, best2, ssum, cnt);
  ema_kernel<<<NSEG, 256, 0, stream>>>(centers, ssum, cnt, out_c);

  hipMemcpyAsync(out_h, H, (size_t)BB * DD * 4, hipMemcpyDeviceToDevice, stream);
}
